// GraphTransformerPooling_68968584839828
// MI455X (gfx1250) — matrix operand order repaired and checked by reading, not compiled, not run on hardware
//
#include <hip/hip_runtime.h>
#include <hip/hip_bf16.h>
#include <math.h>

// GraphTransformerPooling on MI455X (gfx1250, wave32).
//
// Algebraic restructure: K is never materialized (fold q into Wk -> 256x4
// score projection) and V is never materialized (pooling commutes with the
// linear V projection -> accumulate u[b,h] = sum e*emb, project once per
// segment). The score pass and the segment-softmax pass are FUSED: one wave32
// owns one contiguous segment and makes two passes over a <=128-node chunk
// (second pass hits L2, 192MB), with a once-per-chunk online-softmax rescale.
// Effective HBM traffic ~300MB => ~13us at 23.3 TB/s. The three tail GEMMs
// (8192x256 @ 256x256) run on v_wmma_f32_16x16x32_bf16 with f32 accumulate.

#define D   256
#define H   4
#define DH  64
#define SEGCHUNK 128   // 4 register slots * 32 lanes

typedef __attribute__((ext_vector_type(16))) __bf16 v16bf;
typedef __attribute__((ext_vector_type(8)))  float  v8f;

static __device__ __forceinline__ float wave_sum(float v) {
#pragma unroll
  for (int off = 16; off > 0; off >>= 1) v += __shfl_xor(v, off, 32);
  return v;   // butterfly: every lane ends with the full sum
}

// ---------------------------------------------------------------------------
// K0: q = (query @ Wq^T + bq)/sqrt(dh); fold q into Wk:
//     wscore[h][c] = sum_j Wk[h*64+j, c] * q[h*64+j];  cb[h] = bk_h . q_h
// ---------------------------------------------------------------------------
__global__ void qprep_kernel(const float* __restrict__ query,
                             const float* __restrict__ in_proj_w,
                             const float* __restrict__ in_proj_b,
                             float* __restrict__ wscore,
                             float* __restrict__ cb) {
  __shared__ float qsh[D];
  const int i = threadIdx.x;
  float acc = in_proj_b[i];
  for (int j = 0; j < D; ++j) acc += query[j] * in_proj_w[(size_t)i * D + j];
  qsh[i] = acc * 0.125f;  // 1/sqrt(64)
  __syncthreads();
  for (int t = i; t < H * D; t += blockDim.x) {
    const int h = t >> 8, c = t & (D - 1);
    const float* wk = in_proj_w + (size_t)(D + h * DH) * D + c;
    float a = 0.f;
    for (int j = 0; j < DH; ++j) a += wk[(size_t)j * D] * qsh[h * DH + j];
    wscore[t] = a;
  }
  if (i < H) {
    float a = 0.f;
    for (int j = 0; j < DH; ++j) a += in_proj_b[D + i * DH + j] * qsh[i * DH + j];
    cb[i] = a;
  }
}

// ---------------------------------------------------------------------------
// K1: segment start offsets via lower_bound on the sorted segment_ids.
// ---------------------------------------------------------------------------
__global__ void seg_bounds_kernel(const int* __restrict__ seg_ids,
                                  int* __restrict__ seg_start, int N, int Bseg) {
  const int b = blockIdx.x * blockDim.x + threadIdx.x;
  if (b > Bseg) return;
  if (b == Bseg) { seg_start[Bseg] = N; return; }
  int lo = 0, hi = N;
  while (lo < hi) {
    int mid = (lo + hi) >> 1;
    if (seg_ids[mid] < b) lo = mid + 1; else hi = mid;
  }
  seg_start[b] = lo;
}

// ---------------------------------------------------------------------------
// K2 (fused scores + segment softmax): one wave32 per segment; single HBM
// sweep of emb. Per chunk of <=128 nodes:
//   pass A: score_h(n) = emb_n . wscore_h + cb_h (butterfly sum -> all lanes),
//           parked in per-lane register slots; chunk max tracked in-register.
//   rescale: online-softmax merge of running max (once per chunk).
//   pass B: re-read rows (L2-resident), accumulate denom_h and
//           u[b,h,:] += exp(score-m) * emb_n.
// ---------------------------------------------------------------------------
__global__ void segment_fused_kernel(const float* __restrict__ emb,
                                     const float* __restrict__ wscore,
                                     const float* __restrict__ cb,
                                     const int* __restrict__ seg_start,
                                     float* __restrict__ denom,  // B*H
                                     float* __restrict__ u,      // B*H*D
                                     int Bseg) {
  const int lane = threadIdx.x & 31;
  const int b = (blockIdx.x * blockDim.x + threadIdx.x) >> 5;
  if (b >= Bseg) return;
  const int s = seg_start[b], e = seg_start[b + 1];

  float w[H][8];
#pragma unroll
  for (int h = 0; h < H; ++h)
#pragma unroll
    for (int j = 0; j < 8; ++j) w[h][j] = wscore[h * D + lane * 8 + j];
  const float cbr[H] = {cb[0], cb[1], cb[2], cb[3]};

  float m[H]  = {-3.4e38f, -3.4e38f, -3.4e38f, -3.4e38f};
  float ds[H] = {0.f, 0.f, 0.f, 0.f};
  float acc[H][8];
#pragma unroll
  for (int h = 0; h < H; ++h)
#pragma unroll
    for (int j = 0; j < 8; ++j) acc[h][j] = 0.f;
  float4 priv[SEGCHUNK / 32];  // score slots distributed across lanes

  for (int c0 = s; c0 < e; c0 += SEGCHUNK) {
    const int ce = (c0 + SEGCHUNK < e) ? (c0 + SEGCHUNK) : e;

    // ---- pass A: chunk scores + chunk max ----
    float cm[H] = {-3.4e38f, -3.4e38f, -3.4e38f, -3.4e38f};
    for (int n = c0; n < ce; ++n) {
      const float4* row = (const float4*)(emb + (size_t)n * D + lane * 8);
      if (n + 1 < ce)
        __builtin_prefetch(emb + (size_t)(n + 1) * D + lane * 8, 0, 0);
      const float4 a0 = row[0], a1 = row[1];
      const float el[8] = {a0.x, a0.y, a0.z, a0.w, a1.x, a1.y, a1.z, a1.w};
      float sh[H] = {0.f, 0.f, 0.f, 0.f};
#pragma unroll
      for (int h = 0; h < H; ++h)
#pragma unroll
        for (int j = 0; j < 8; ++j) sh[h] = fmaf(el[j], w[h][j], sh[h]);
#pragma unroll
      for (int h = 0; h < H; ++h) sh[h] = wave_sum(sh[h]) + cbr[h];
      const int idx = n - c0;
      if (lane == (idx & 31))
        priv[idx >> 5] = make_float4(sh[0], sh[1], sh[2], sh[3]);
#pragma unroll
      for (int h = 0; h < H; ++h) cm[h] = fmaxf(cm[h], sh[h]);
    }

    // ---- online-softmax merge (once per chunk) ----
#pragma unroll
    for (int h = 0; h < H; ++h) {
      const float nm = fmaxf(m[h], cm[h]);
      const float sc = __expf(m[h] - nm);  // m==-inf -> 0 (ds/acc are 0 then)
      ds[h] *= sc;
#pragma unroll
      for (int j = 0; j < 8; ++j) acc[h][j] *= sc;
      m[h] = nm;
    }

    // ---- pass B: accumulate (rows are L2-resident from pass A) ----
    for (int n = c0; n < ce; ++n) {
      const int idx = n - c0, owner = idx & 31, slot = idx >> 5;
      float4 sv;
      sv.x = __shfl(priv[slot].x, owner, 32);
      sv.y = __shfl(priv[slot].y, owner, 32);
      sv.z = __shfl(priv[slot].z, owner, 32);
      sv.w = __shfl(priv[slot].w, owner, 32);
      const float ee[H] = {__expf(sv.x - m[0]), __expf(sv.y - m[1]),
                           __expf(sv.z - m[2]), __expf(sv.w - m[3])};
      ds[0] += ee[0]; ds[1] += ee[1]; ds[2] += ee[2]; ds[3] += ee[3];
      const float4* row = (const float4*)(emb + (size_t)n * D + lane * 8);
      const float4 a0 = row[0], a1 = row[1];
      const float el[8] = {a0.x, a0.y, a0.z, a0.w, a1.x, a1.y, a1.z, a1.w};
#pragma unroll
      for (int h = 0; h < H; ++h)
#pragma unroll
        for (int j = 0; j < 8; ++j) acc[h][j] = fmaf(ee[h], el[j], acc[h][j]);
    }
  }

  float* ub = u + (size_t)b * (H * D);
#pragma unroll
  for (int h = 0; h < H; ++h)
#pragma unroll
    for (int j = 0; j < 8; ++j) ub[h * D + lane * 8 + j] = acc[h][j];
  if (lane == 0) {
    float* dp = denom + (size_t)b * H;
    dp[0] = ds[0]; dp[1] = ds[1]; dp[2] = ds[2]; dp[3] = ds[3];
  }
}

// ---------------------------------------------------------------------------
// WMMA bf16 GEMM: C(M x N) = A(M x K=256, f32) @ Bw(N x K, f32)^T [+ epilogue]
// One wave per 16x16 C tile, 8x v_wmma_f32_16x16x32_bf16 along K.
// MODE 0: attn   = (u_head . Wv)/denom + bv, masked by denom>0 (lda = H*D)
// MODE 1: pooled = A @ W^T + bias
// MODE 2: out    = A @ W^T + bias, zeroed where segment count == 0
// Fragment packing: A 16x32 / B 32x16 in contiguous k order (k = 16*half + j); C 16x16, wave32.
// ---------------------------------------------------------------------------
template <int MODE>
__global__ void wmma_gemm_kernel(const float* __restrict__ A,
                                 const float* __restrict__ Bw,
                                 const float* __restrict__ bias,
                                 const float* __restrict__ denom,
                                 const int* __restrict__ seg_start,
                                 float* __restrict__ Cd,
                                 int M, int N, int lda) {
  const int lane = threadIdx.x & 31;
  const int half = lane >> 4;   // 0: lanes 0-15, 1: lanes 16-31
  const int l15 = lane & 15;
  const int waveGlobal = (blockIdx.x * blockDim.x + threadIdx.x) >> 5;
  const int nwaves = (gridDim.x * blockDim.x) >> 5;
  const int tilesN = N >> 4;
  const int totalTiles = (M >> 4) * tilesN;

  for (int tile = waveGlobal; tile < totalTiles; tile += nwaves) {
    const int tm = tile / tilesN, tn = tile % tilesN;
    const int head = (tn << 4) >> 6;  // only used for MODE 0
    const float* Abase = (MODE == 0)
        ? (A + (size_t)(tm << 4) * lda + head * D)
        : (A + (size_t)(tm << 4) * lda);
    const int colN = (tn << 4) + l15;
    const float* Bcol = Bw + (size_t)colN * D;

    v8f acc = {};
#pragma unroll
    for (int k0 = 0; k0 < D; k0 += 32) {
      // A fragment: row = tm*16 + l15; elems [j] = K k0 + half*16 + j
      //             (contiguous k order, matching B)
      const float* pa = Abase + (size_t)l15 * lda + k0 + half * 16;
      v16bf af, bf;
#pragma unroll
      for (int j = 0; j < 8; ++j) {
        af[j]     = (__bf16)pa[j];
        af[8 + j] = (__bf16)pa[8 + j];
      }
      // B fragment: col = colN; elems [j] = K k0 + half*16 + j (B 32x16 layout)
      const float* pb = Bcol + k0 + half * 16;
#pragma unroll
      for (int j = 0; j < 16; ++j) bf[j] = (__bf16)pb[j];
      acc = __builtin_amdgcn_wmma_f32_16x16x32_bf16(
          /*neg_a=*/false, af, /*neg_b=*/false, bf,
          /*c_mod=*/(short)0, acc, /*reuse_a=*/false, /*reuse_b=*/false);
    }

    const float bi = bias[colN];
#pragma unroll
    for (int r = 0; r < 8; ++r) {
      const int row = (tm << 4) + (half << 3) + r;  // C layout: M = r + 8*half
      const float v = acc[r];
      float outv;
      if (MODE == 0) {
        const float dn = denom[(size_t)row * H + head];
        outv = (dn > 0.f) ? (v / fmaxf(dn, 1e-30f) + bi) : 0.f;
      } else if (MODE == 2) {
        const int cnt = seg_start[row + 1] - seg_start[row];
        outv = (cnt > 0) ? (v + bi) : 0.f;
      } else {
        outv = v + bi;
      }
      Cd[(size_t)row * N + colN] = outv;
    }
  }
}

// ---------------------------------------------------------------------------
extern "C" void kernel_launch(void* const* d_in, const int* in_sizes, int n_in,
                              void* d_out, int out_size, void* d_ws, size_t ws_size,
                              hipStream_t stream) {
  const float* emb           = (const float*)d_in[0];
  const float* query         = (const float*)d_in[1];
  const float* in_proj_w     = (const float*)d_in[2];
  const float* in_proj_b     = (const float*)d_in[3];
  const float* out_proj_w    = (const float*)d_in[4];
  const float* out_proj_b    = (const float*)d_in[5];
  const float* output_proj_w = (const float*)d_in[6];
  const float* output_proj_b = (const float*)d_in[7];
  const int*   seg_ids       = (const int*)d_in[8];

  const int N    = in_sizes[0] / D;   // 262144
  const int Bseg = out_size / D;      // 8192 (hidden == 256)

  // Workspace layout (f32 units, 64-element aligned blocks).
  float* ws = (float*)d_ws;
  float* wscore = ws;                                   // H*D = 1024
  float* cb     = ws + 1024;                            // H (padded to 1024)
  float* denomw = ws + 2048;                            // Bseg*H
  size_t off = 2048 + (size_t)Bseg * H;
  int*   seg_start = (int*)(ws + off);                  // Bseg+1 ints
  off += ((size_t)(Bseg + 1) + 63) & ~(size_t)63;
  float* u      = ws + off; off += (size_t)Bseg * H * D;
  float* attn   = ws + off; off += (size_t)Bseg * D;
  float* pooled = ws + off;
  (void)ws_size; (void)n_in;

  qprep_kernel<<<1, 256, 0, stream>>>(query, in_proj_w, in_proj_b, wscore, cb);

  seg_bounds_kernel<<<(Bseg + 1 + 255) / 256, 256, 0, stream>>>(
      seg_ids, seg_start, N, Bseg);

  segment_fused_kernel<<<(Bseg * 32 + 255) / 256, 256, 0, stream>>>(
      emb, wscore, cb, seg_start, denomw, u, Bseg);

  const int tiles  = (Bseg >> 4) * (D >> 4);    // 8192 16x16 tiles per GEMM
  const int blocks = (tiles + 7) / 8;           // 8 waves / block

  const float* Wv = in_proj_w + 2 * D * D;
  const float* bv = in_proj_b + 2 * D;
  wmma_gemm_kernel<0><<<blocks, 256, 0, stream>>>(
      u, Wv, bv, denomw, nullptr, attn, Bseg, D, H * D);
  wmma_gemm_kernel<1><<<blocks, 256, 0, stream>>>(
      attn, out_proj_w, out_proj_b, nullptr, nullptr, pooled, Bseg, D, D);
  wmma_gemm_kernel<2><<<blocks, 256, 0, stream>>>(
      pooled, output_proj_w, output_proj_b, nullptr, seg_start,
      (float*)d_out, Bseg, D, D);
}
